// DoubleEmbedding_33157147525313
// MI455X (gfx1250) — matrix operand order repaired and checked by reading, not compiled, not run on hardware
//
#include <hip/hip_runtime.h>
#include <hip/hip_bf16.h>

typedef __attribute__((ext_vector_type(16))) _Float16 v16h;
typedef __attribute__((ext_vector_type(8)))  _Float16 v8h;
typedef __attribute__((ext_vector_type(8)))  float    v8f;

#define NN     8192
#define DD     128
#define TOPKK  30
#define ALPHAF 3.0f
#define A_OFF  (3 * NN * TOPKK)   // 737280 floats before A_masked region

// ------------------------------------------------------------------
// Kernel 1: M = tanh(alpha * (E @ W^T + b)), stored as f16 (WMMA operand)
// one block per row, one thread per output column
// ------------------------------------------------------------------
__global__ __launch_bounds__(128) void layer_kernel(
    const float* __restrict__ E, const float* __restrict__ W,
    const float* __restrict__ b, _Float16* __restrict__ M)
{
  __shared__ float se[DD];
  const int row = blockIdx.x;
  const int j   = threadIdx.x;
  se[j] = E[(size_t)row * DD + j];
  __syncthreads();
  const float* wr = W + (size_t)j * DD;   // (E @ W^T)[i,j] = dot(E[i,:], W[j,:])
  float acc = 0.f;
  #pragma unroll 8
  for (int k = 0; k < DD; ++k) acc = fmaf(wr[k], se[k], acc);
  M[(size_t)row * DD + j] = (_Float16)tanhf(ALPHAF * (acc + b[j]));
}

// ------------------------------------------------------------------
// Kernel 2: A = relu(M1@M2^T - M2@M1^T), diag = 0, written once to d_out.
// 256 threads = 8 waves per block; block tile = 128(M) x 64(N).
// Wave w owns rows [w*16, w*16+16) of the tile and all 4 N-subtiles.
// K = 128 staged in LDS in 4 chunks of 32; v_wmma_f32_16x16x32_f16.
// ------------------------------------------------------------------
__global__ __launch_bounds__(256) void score_kernel(
    const _Float16* __restrict__ M1, const _Float16* __restrict__ M2,
    float* __restrict__ Aout)
{
  constexpr int RS = 40;                    // padded LDS row stride (halfs) -> conflict-free
  __shared__ _Float16 sA1[128 * RS];        // M1 rows tileM..tileM+127, K chunk
  __shared__ _Float16 sA2[128 * RS];        // M2 rows tileM..
  __shared__ _Float16 sB1[64 * RS];         // M1 rows tileN..tileN+63  (B of M1^T)
  __shared__ _Float16 sB2[64 * RS];         // M2 rows tileN..          (B of M2^T)

  const int tid   = threadIdx.x;
  const int wv    = tid >> 5;
  const int lane  = tid & 31;
  const int r     = lane & 15;
  const int hi    = lane >> 4;
  const int tileM = blockIdx.y * 128;
  const int tileN = blockIdx.x * 64;

  v8f acc1[4], acc2[4];
  #pragma unroll
  for (int n = 0; n < 4; ++n) {
    acc1[n] = v8f{0.f,0.f,0.f,0.f,0.f,0.f,0.f,0.f};
    acc2[n] = v8f{0.f,0.f,0.f,0.f,0.f,0.f,0.f,0.f};
  }

  for (int kk = 0; kk < 4; ++kk) {
    if (kk) __syncthreads();                // protect LDS reuse across chunks
    const int kb = kk * 32;

    // stage A-side: 128 rows x 32 halfs x 2 mats = 1024 b128 segments / 256 thr
    #pragma unroll
    for (int i = 0; i < 2; ++i) {
      const int seg = tid + i * 256;
      const int row = seg >> 2, sc = seg & 3;
      *(v8h*)&sA1[row * RS + sc * 8] =
          *(const v8h*)&M1[(size_t)(tileM + row) * DD + kb + sc * 8];
      *(v8h*)&sA2[row * RS + sc * 8] =
          *(const v8h*)&M2[(size_t)(tileM + row) * DD + kb + sc * 8];
    }
    // stage B-side: 64 rows x 32 halfs x 2 mats = 512 segments / 256 thr
    {
      const int row = tid >> 2, sc = tid & 3;
      *(v8h*)&sB1[row * RS + sc * 8] =
          *(const v8h*)&M1[(size_t)(tileN + row) * DD + kb + sc * 8];
      *(v8h*)&sB2[row * RS + sc * 8] =
          *(const v8h*)&M2[(size_t)(tileN + row) * DD + kb + sc * 8];
    }
    __syncthreads();

    // A fragments (ISA 16-bit A layout: lo-lanes K{0..7,16..23}, hi-lanes K{8..15,24..31})
    const int abase = (wv * 16 + r) * RS + hi * 8;
    v8h a1lo = *(const v8h*)&sA1[abase];
    v8h a1hi = *(const v8h*)&sA1[abase + 16];
    v8h a2lo = *(const v8h*)&sA2[abase];
    v8h a2hi = *(const v8h*)&sA2[abase + 16];
    v16h a1 = __builtin_shufflevector(a1lo, a1hi, 0,1,2,3,4,5,6,7,8,9,10,11,12,13,14,15);
    v16h a2 = __builtin_shufflevector(a2lo, a2hi, 0,1,2,3,4,5,6,7,8,9,10,11,12,13,14,15);

    #pragma unroll
    for (int n = 0; n < 4; ++n) {
      // B fragments (col=r; lo-lanes K{0..7,16..23}, hi-lanes K{8..15,24..31})
      const int bbase = (n * 16 + r) * RS + hi * 8;
      v8h b2lo = *(const v8h*)&sB2[bbase];
      v8h b2hi = *(const v8h*)&sB2[bbase + 16];
      v8h b1lo = *(const v8h*)&sB1[bbase];
      v8h b1hi = *(const v8h*)&sB1[bbase + 16];
      v16h b2 = __builtin_shufflevector(b2lo, b2hi, 0,1,2,3,4,5,6,7,8,9,10,11,12,13,14,15);
      v16h b1 = __builtin_shufflevector(b1lo, b1hi, 0,1,2,3,4,5,6,7,8,9,10,11,12,13,14,15);
      // acc1 += M1_tile @ M2_tile^T ; acc2 += M2_tile @ M1_tile^T
      acc1[n] = __builtin_amdgcn_wmma_f32_16x16x32_f16(
          false, a1, false, b2, (short)0, acc1[n], false, false);
      acc2[n] = __builtin_amdgcn_wmma_f32_16x16x32_f16(
          false, a2, false, b1, (short)0, acc2[n], false, false);
    }
  }

  // epilogue: relu(antisymmetric), zero diagonal, single write of A
  #pragma unroll
  for (int n = 0; n < 4; ++n) {
    const int col = tileN + n * 16 + r;
    #pragma unroll
    for (int j = 0; j < 8; ++j) {
      const int row = tileM + wv * 16 + j + 8 * hi;   // C layout: lo-lanes M=j, hi-lanes M=j+8
      float v = acc1[n][j] - acc2[n][j];
      v = v > 0.f ? v : 0.f;
      if (row == col) v = 0.f;
      Aout[(size_t)row * NN + col] = v;
    }
  }
}

// ------------------------------------------------------------------
// Kernel 3: per-row top-30 (stable, lowest-index tie-break like lax.top_k),
// rewrite row as sparse mask in place, emit (src, dst, attr) edge arrays.
// One 256-thread block per row; row cached in LDS.
// ------------------------------------------------------------------
__global__ __launch_bounds__(256) void topk_kernel(float* __restrict__ out)
{
  __shared__ float rowv[NN];
  __shared__ float redv[256];
  __shared__ int   redi[256];
  __shared__ float topv[TOPKK];
  __shared__ int   topi[TOPKK];

  const int tid = threadIdx.x;
  const int row = blockIdx.x;
  float* Arow = out + A_OFF + (size_t)row * NN;

  for (int i = tid; i < NN / 4; i += 256)
    ((float4*)rowv)[i] = ((const float4*)Arow)[i];
  __syncthreads();

  for (int t = 0; t < TOPKK; ++t) {
    // local argmax over a contiguous 32-chunk (strict > keeps earliest index)
    const int base = tid * 32;
    float bv = rowv[base];
    int   bi = base;
    #pragma unroll
    for (int k = 1; k < 32; ++k) {
      float v = rowv[base + k];
      if (v > bv) { bv = v; bi = base + k; }
    }
    redv[tid] = bv; redi[tid] = bi;
    __syncthreads();
    for (int s = 128; s > 0; s >>= 1) {
      if (tid < s) {
        float ov = redv[tid + s]; int oi = redi[tid + s];
        if (ov > redv[tid] || (ov == redv[tid] && oi < redi[tid])) {
          redv[tid] = ov; redi[tid] = oi;
        }
      }
      __syncthreads();
    }
    if (tid == 0) {
      topv[t] = redv[0];
      topi[t] = redi[0];
      rowv[redi[0]] = -1.0f;    // A >= 0, so -1 is a safe "removed" sentinel
    }
    __syncthreads();
  }

  // zero the row, then scatter the kept top-k values
  float4 z; z.x = z.y = z.z = z.w = 0.f;
  for (int i = tid; i < NN / 4; i += 256) ((float4*)Arow)[i] = z;
  __syncthreads();
  if (tid < TOPKK) {
    const int e = row * TOPKK + tid;
    Arow[topi[tid]]          = topv[tid];
    out[e]                   = (float)row;        // edge_indices[0] (src)
    out[NN * TOPKK + e]      = (float)topi[tid];  // edge_indices[1] (dst)
    out[2 * NN * TOPKK + e]  = topv[tid];         // edge_attr
  }
}

// ------------------------------------------------------------------
extern "C" void kernel_launch(void* const* d_in, const int* in_sizes, int n_in,
                              void* d_out, int out_size, void* d_ws, size_t ws_size,
                              hipStream_t stream)
{
  (void)in_sizes; (void)n_in; (void)out_size; (void)ws_size;
  const float* E1 = (const float*)d_in[0];
  const float* E2 = (const float*)d_in[1];
  const float* W1 = (const float*)d_in[2];
  const float* b1 = (const float*)d_in[3];
  const float* W2 = (const float*)d_in[4];
  const float* b2 = (const float*)d_in[5];
  float* out = (float*)d_out;

  _Float16* M1h = (_Float16*)d_ws;                 // 2 MB
  _Float16* M2h = M1h + (size_t)NN * DD;           // 2 MB

  layer_kernel<<<NN, DD, 0, stream>>>(E1, W1, b1, M1h);
  layer_kernel<<<NN, DD, 0, stream>>>(E2, W2, b2, M2h);

  dim3 g2(NN / 64, NN / 128);                      // 128 x 64 = 8192 blocks
  score_kernel<<<g2, 256, 0, stream>>>(M1h, M2h, out + A_OFF);

  topk_kernel<<<NN, 256, 0, stream>>>(out);
}
